// MultiHeadAttention_35940286333141
// MI455X (gfx1250) — matrix operand order repaired and checked by reading, not compiled, not run on hardware
//
#include <hip/hip_runtime.h>
#include <hip/hip_bf16.h>

#define BATCH   4
#define SEQ     2048
#define DMODEL  1024
#define NHEADS  16
#define HDIM    64
#define QROWS   128              // q rows per block (16 per wave * 8 waves)
#define KTILE   32               // keys per iteration (= WMMA K for bf16)
#define ATT_SCALE 0.125f         // 1/sqrt(64)

typedef __attribute__((ext_vector_type(16))) __bf16 v16bf;
typedef __attribute__((ext_vector_type(8)))  float  v8f;

union Pack16 { uint4 u[2]; v16bf v; };
union Pack8  { uint4 u; __bf16 b[8]; };

// Load a 16-element bf16 fragment as two 16B chunks (maps to ds_load_b128 /
// global_load_b128) from two base pointers (each 8 contiguous bf16).
static __device__ inline v16bf load8x2(const __bf16* p0, const __bf16* p1) {
    Pack16 pk;
    pk.u[0] = *reinterpret_cast<const uint4*>(p0);
    pk.u[1] = *reinterpret_cast<const uint4*>(p1);
    return pk.v;
}

// Build a 16x32 bf16 A-fragment slice from a global f32 row:
// elements 0..7  = row[d0..d0+7], elements 8..15 = row[d1..d1+7]
static __device__ inline v16bf load_a_frag_f32(const float* row, int d0, int d1) {
    float4 a0 = *reinterpret_cast<const float4*>(row + d0);
    float4 a1 = *reinterpret_cast<const float4*>(row + d0 + 4);
    float4 b0 = *reinterpret_cast<const float4*>(row + d1);
    float4 b1 = *reinterpret_cast<const float4*>(row + d1 + 4);
    v16bf r;
    r[0]=(__bf16)a0.x;  r[1]=(__bf16)a0.y;  r[2]=(__bf16)a0.z;  r[3]=(__bf16)a0.w;
    r[4]=(__bf16)a1.x;  r[5]=(__bf16)a1.y;  r[6]=(__bf16)a1.z;  r[7]=(__bf16)a1.w;
    r[8]=(__bf16)b0.x;  r[9]=(__bf16)b0.y;  r[10]=(__bf16)b0.z; r[11]=(__bf16)b0.w;
    r[12]=(__bf16)b1.x; r[13]=(__bf16)b1.y; r[14]=(__bf16)b1.z; r[15]=(__bf16)b1.w;
    return r;
}

__global__ __launch_bounds__(256)
void MultiHeadAttention_35940286333141_kernel(
        const float* __restrict__ Q, const float* __restrict__ K,
        const float* __restrict__ V, const float* __restrict__ mask,
        const float* __restrict__ relpos, float* __restrict__ out)
{
    __shared__ __bf16 sK[KTILE][HDIM];      // key-major:   [key][d]    4KB
    __shared__ __bf16 sVT[HDIM][KTILE];     // transposed:  [d][key]    4KB
    __shared__ __bf16 sP[8][16][KTILE];     // per-wave P:  [w][m][k]   8KB

    const int tid  = threadIdx.x;
    const int wave = tid >> 5;
    const int lane = tid & 31;
    const int hi   = lane >> 4;     // lane half (WMMA layout)
    const int ln   = lane & 15;

    // ---- L2-locality grid decode: qt outermost so the 64 blocks of one
    // q-supertile run together.  Their combined working set (mask 4MB +
    // relpos 16MB + ALL of K/V 64MB) fits in the 192MB L2, giving full
    // cross-head mask reuse, cross-batch relpos reuse, and cross-qt K/V
    // reuse -> HBM traffic drops to the unique ~448MB.
    const int blk = blockIdx.x;
    const int qt  = blk / (BATCH * NHEADS);
    const int b   = (blk / NHEADS) % BATCH;
    const int h   = blk % NHEADS;
    const int qbase = qt * QROWS + wave * 16;

    // ---- Q A-fragments (16x32 bf16 each; A-layout: lane row = ln,
    //      lanes 0-15 hold K=0..15; lanes 16-31 hold K=16..31)
    const float* qrow = Q + ((size_t)(b * SEQ + qbase + ln) * DMODEL) + h * HDIM;
    v16bf aq0 = load_a_frag_f32(qrow, 0  + hi * 16, 8 + hi * 16);   // d 0..31
    v16bf aq1 = load_a_frag_f32(qrow, 32 + hi * 16, 40 + hi * 16);   // d 32..63

    v8f oacc[4] = {v8f{}, v8f{}, v8f{}, v8f{}};
    float mrow[8], lrow[8];
    #pragma unroll
    for (int v = 0; v < 8; ++v) { mrow[v] = -3.0e38f; lrow[v] = 0.0f; }

    const size_t mask_base = (size_t)b * SEQ * SEQ;
    const size_t rel_base  = (size_t)h * SEQ * SEQ;

    for (int kb = 0; kb < SEQ; kb += KTILE) {
        __syncthreads();   // previous iteration done reading sK/sVT

        // ---- cooperative staging: K tile (row-major) + V tile (transposed)
        {
            const int key  = tid >> 3;         // 0..31
            const int dseg = (tid & 7) * 8;    // 0,8,...,56
            const size_t rowoff = ((size_t)(b * SEQ + kb + key) * DMODEL) + h * HDIM + dseg;

            const float* kp = K + rowoff;
            float4 k0 = *reinterpret_cast<const float4*>(kp);
            float4 k1 = *reinterpret_cast<const float4*>(kp + 4);
            Pack8 pk;
            pk.b[0]=(__bf16)k0.x; pk.b[1]=(__bf16)k0.y; pk.b[2]=(__bf16)k0.z; pk.b[3]=(__bf16)k0.w;
            pk.b[4]=(__bf16)k1.x; pk.b[5]=(__bf16)k1.y; pk.b[6]=(__bf16)k1.z; pk.b[7]=(__bf16)k1.w;
            *reinterpret_cast<uint4*>(&sK[key][dseg]) = pk.u;

            const float* vp = V + rowoff;
            float4 v0 = *reinterpret_cast<const float4*>(vp);
            float4 v1 = *reinterpret_cast<const float4*>(vp + 4);
            const float vv[8] = {v0.x, v0.y, v0.z, v0.w, v1.x, v1.y, v1.z, v1.w};
            #pragma unroll
            for (int j = 0; j < 8; ++j) sVT[dseg + j][key] = (__bf16)vv[j];

            if (kb + KTILE < SEQ) {   // prefetch next key tile
                __builtin_prefetch(kp + (size_t)KTILE * DMODEL, 0, 1);
                __builtin_prefetch(vp + (size_t)KTILE * DMODEL, 0, 1);
            }
        }
        __syncthreads();

        // ---- S = Q K^T  (two 16x16 f32 tiles; key sub-tiles t=0,1)
        v8f s0c = {}, s1c = {};
        {
            const __bf16* kr0 = &sK[ln][0];        // B col n = key ln        (t=0)
            const __bf16* kr1 = &sK[16 + ln][0];   //                         (t=1)
            // B-layout: lanes 0-15 hold K=0..15, lanes 16-31 K=16..31 (contiguous d)
            v16bf bk00 = load8x2(kr0 + hi * 16,      kr0 + hi * 16 + 8);       // d chunk 0
            v16bf bk01 = load8x2(kr0 + 32 + hi * 16, kr0 + 32 + hi * 16 + 8);  // d chunk 1
            v16bf bk10 = load8x2(kr1 + hi * 16,      kr1 + hi * 16 + 8);
            v16bf bk11 = load8x2(kr1 + 32 + hi * 16, kr1 + 32 + hi * 16 + 8);
            s0c = __builtin_amdgcn_wmma_f32_16x16x32_bf16(false, aq0, false, bk00, (short)0, s0c, false, false);
            s0c = __builtin_amdgcn_wmma_f32_16x16x32_bf16(false, aq1, false, bk01, (short)0, s0c, false, false);
            s1c = __builtin_amdgcn_wmma_f32_16x16x32_bf16(false, aq0, false, bk10, (short)0, s1c, false, false);
            s1c = __builtin_amdgcn_wmma_f32_16x16x32_bf16(false, aq1, false, bk11, (short)0, s1c, false, false);
        }

        // ---- bias: scores = (S - mask + relpos) * scale
        // C-layout: VGPR v <-> row m = v + 8*hi, col n = ln (+16 for tile 1)
        float s0[8], s1[8];
        #pragma unroll
        for (int v = 0; v < 8; ++v) {
            const int m = v + hi * 8;
            const size_t ro = (size_t)(qbase + m) * SEQ + kb;
            const float mk0 = mask[mask_base + ro + ln];
            const float mk1 = mask[mask_base + ro + 16 + ln];
            const float rl0 = relpos[rel_base + ro + ln];
            const float rl1 = relpos[rel_base + ro + 16 + ln];
            s0[v] = (s0c[v] - mk0 + rl0) * ATT_SCALE;
            s1[v] = (s1c[v] - mk1 + rl1) * ATT_SCALE;
        }

        // ---- online softmax (row stats per VGPR slot; 16-lane reductions)
        #pragma unroll
        for (int v = 0; v < 8; ++v) {
            float tm = fmaxf(s0[v], s1[v]);
            #pragma unroll
            for (int off = 1; off < 16; off <<= 1)
                tm = fmaxf(tm, __shfl_xor(tm, off, 32));
            const float mn = fmaxf(mrow[v], tm);
            const float alpha = __expf(mrow[v] - mn);
            mrow[v] = mn;
            const float p0 = __expf(s0[v] - mn);
            const float p1 = __expf(s1[v] - mn);
            s0[v] = p0; s1[v] = p1;
            float rs = p0 + p1;
            #pragma unroll
            for (int off = 1; off < 16; off <<= 1)
                rs += __shfl_xor(rs, off, 32);
            lrow[v] = lrow[v] * alpha + rs;
            oacc[0][v] *= alpha; oacc[1][v] *= alpha;
            oacc[2][v] *= alpha; oacc[3][v] *= alpha;
        }

        // ---- write P (C-layout) to per-wave LDS as bf16
        #pragma unroll
        for (int v = 0; v < 8; ++v) {
            const int m = v + hi * 8;
            sP[wave][m][ln]      = (__bf16)s0[v];
            sP[wave][m][16 + ln] = (__bf16)s1[v];
        }
        __syncthreads();   // P visible across lanes (and staging fence)

        // ---- O += P V : reload P as A-fragment, V^T rows as B-fragments
        const __bf16* prow = &sP[wave][ln][0];
        const v16bf pa = load8x2(prow + hi * 16, prow + hi * 16 + 8);
        #pragma unroll
        for (int c = 0; c < 4; ++c) {
            const __bf16* vr = &sVT[c * 16 + ln][hi * 16];
            const v16bf bv = load8x2(vr, vr + 8);
            oacc[c] = __builtin_amdgcn_wmma_f32_16x16x32_bf16(false, pa, false, bv, (short)0, oacc[c], false, false);
        }
    }

    // ---- normalize and write out: out[b][q][h*64 + c*16 + ln]
    float* orow0 = out + ((size_t)(b * SEQ + qbase) * DMODEL) + h * HDIM + ln;
    #pragma unroll
    for (int v = 0; v < 8; ++v) {
        const int m = v + hi * 8;
        const float inv = 1.0f / lrow[v];
        float* op = orow0 + (size_t)m * DMODEL;
        op[0]  = oacc[0][v] * inv;
        op[16] = oacc[1][v] * inv;
        op[32] = oacc[2][v] * inv;
        op[48] = oacc[3][v] * inv;
    }
}

extern "C" void kernel_launch(void* const* d_in, const int* in_sizes, int n_in,
                              void* d_out, int out_size, void* d_ws, size_t ws_size,
                              hipStream_t stream) {
    const float* Q      = (const float*)d_in[0];
    const float* K      = (const float*)d_in[1];
    const float* V      = (const float*)d_in[2];
    const float* mask   = (const float*)d_in[3];
    const float* relpos = (const float*)d_in[4];
    float* out = (float*)d_out;

    const int grid = BATCH * NHEADS * (SEQ / QROWS);   // 1024 blocks
    MultiHeadAttention_35940286333141_kernel<<<grid, 256, 0, stream>>>(
        Q, K, V, mask, relpos, out);
}
